// TTTGating_7679401525797
// MI455X (gfx1250) — compile-verified, matrix operand order checked by reading, not run on hardware
//
#include <hip/hip_runtime.h>
#include <hip/hip_bf16.h>
#include <math.h>

// ---------------------------------------------------------------------------
// TTT-style gated fast-weight layer for MI455X (gfx1250, wave32, WMMA).
// All heavy GEMMs use v_wmma_f32_16x16x32_bf16 (bf16 in, f32 accumulate).
// Conv(K=2)+W_target folded into two DxD matrices so that stage is one GEMM.
// Staging uses GLOBAL_LOAD_ASYNC_TO_LDS_B128 (ASYNCcnt) with double-buffered
// LDS: one barrier per K-step, async fill of buffer p^1 overlaps WMMA on p.
// ---------------------------------------------------------------------------

#define DIMK    1024
#define HIDDENK 2688
#define HID2K   5376
#define CHUNKK  256
#define BATCHK  4
#define SEQK    4096
#define ROWSK   (BATCHK * SEQK)   // 16384
#define NCHUNKK (SEQK / CHUNKK)   // 16

#define LR_C   1e-3f
#define CLIP_C 1e-5f
#define EPS_C  1e-12f

typedef __bf16 bf16;
typedef __attribute__((ext_vector_type(16))) __bf16 v16bf;
typedef __attribute__((ext_vector_type(8)))  __bf16 v8bf;
typedef __attribute__((ext_vector_type(4)))  __bf16 v4bf;
typedef __attribute__((ext_vector_type(8)))  float  v8f;

#define BKT 32
#define LP 40   // LDS row pitch (bf16 elems): 80B rows, 16B aligned, conflict-light

__device__ __forceinline__ bf16 f2bf(float f) {
  unsigned u = __builtin_bit_cast(unsigned, f);
  unsigned r = u + 0x7FFFu + ((u >> 16) & 1u);     // round-to-nearest-even
  return __builtin_bit_cast(bf16, (unsigned short)(r >> 16));
}

__device__ __forceinline__ v8bf bfzero8() {
  v8bf z;
#pragma unroll
  for (int j = 0; j < 8; ++j) z[j] = __builtin_bit_cast(bf16, (unsigned short)0);
  return z;
}

__device__ __forceinline__ v8f wmma_bf16(v16bf a, v16bf b, v8f c) {
  return __builtin_amdgcn_wmma_f32_16x16x32_bf16(false, a, false, b, (short)0, c,
                                                 false, false);
}

// Low 32 bits of a generic shared pointer = LDS byte offset (ISA 10.2:
// flat->LDS mapping truncates to addr[31:0]).
__device__ __forceinline__ unsigned lds_off(const void* p) {
  return (unsigned)(unsigned long long)p;
}

// Async direct-to-LDS 16B/lane copy, tracked by ASYNCcnt (ISA 15.18 op 98).
__device__ __forceinline__ void async_ld_b128(unsigned ldsaddr, const void* gaddr) {
  asm volatile("global_load_async_to_lds_b128 %0, %1, off"
               :: "v"(ldsaddr), "v"(gaddr)
               : "memory");
}
__device__ __forceinline__ void wait_asynccnt0() {
  asm volatile("s_wait_asynccnt 0" ::: "memory");
}

// Gather one 16x32 (rows x K) operand fragment from an LDS tile (pitch LP).
// lane<16 row holds K 0-7 (lo) and 16-23 (hi); lane>=16: K 8-15 / 24-31.
__device__ __forceinline__ v16bf lds_frag(const bf16* t, int row0, int lane) {
  int r  = row0 + (lane & 15);
  int kb = (lane >> 4) << 3;                 // 0 or 8
  const bf16* p = t + r * LP + kb;
  union { v16bf v; struct { v8bf lo, hi; } s; } u;
  u.s.lo = *(const v8bf*)(p);                // K = kb .. kb+7
  u.s.hi = *(const v8bf*)(p + 16);           // K = kb+16 .. kb+23
  return u.v;
}

// --------------------------- elementwise helpers ---------------------------

__global__ void k_cvt_bf16(const float* __restrict__ src, bf16* __restrict__ dst,
                           int n4) {
  int stride = gridDim.x * blockDim.x;
  for (int i = blockIdx.x * blockDim.x + threadIdx.x; i < n4; i += stride) {
    float4 f = ((const float4*)src)[i];
    v4bf o;
    o[0] = f2bf(f.x); o[1] = f2bf(f.y); o[2] = f2bf(f.z); o[3] = f2bf(f.w);
    ((v4bf*)dst)[i] = o;
  }
}

// M_j = W_target @ conv_w[:,:,j]  (DxD f32 dots -> bf16), j = blockIdx.z
__global__ __launch_bounds__(256) void k_fold(const float* __restrict__ Wt,
                                              const float* __restrict__ conv,
                                              bf16* __restrict__ M01) {
  __shared__ float Ws[16][17];
  __shared__ float Cs[16][17];
  int tx = threadIdx.x, ty = threadIdx.y;
  int d = blockIdx.y * 16 + ty;
  int c = blockIdx.x * 16 + tx;
  int j = blockIdx.z;
  float acc = 0.f;
  for (int o0 = 0; o0 < DIMK; o0 += 16) {
    Ws[ty][tx] = Wt[(size_t)d * DIMK + o0 + tx];
    Cs[ty][tx] = conv[((size_t)(o0 + ty) * DIMK + c) * 2 + j];
    __syncthreads();
#pragma unroll
    for (int p = 0; p < 16; ++p) acc += Ws[ty][p] * Cs[p][tx];
    __syncthreads();
  }
  M01[((size_t)j * DIMK + d) * DIMK + c] = f2bf(acc);
}

// per-batch copies of w_down (f32 + bf16); batch = blockIdx.y
__global__ void k_init_w(const float* __restrict__ wdn, float* __restrict__ Wf,
                         bf16* __restrict__ Wb) {
  const size_t per = (size_t)DIMK * HIDDENK;
  size_t base = (size_t)blockIdx.y * per;
  size_t stride = (size_t)gridDim.x * blockDim.x;
  for (size_t i = blockIdx.x * (size_t)blockDim.x + threadIdx.x; i < per; i += stride) {
    float w = wdn[i];
    Wf[base + i] = w;
    Wb[base + i] = f2bf(w);
  }
}

// clipped fast-weight update; batch = blockIdx.y
__global__ void k_update(const float* __restrict__ DELTA,
                         const float* __restrict__ nAcc,
                         float* __restrict__ Wf, bf16* __restrict__ Wb) {
  const size_t per = (size_t)DIMK * HIDDENK;
  int b = blockIdx.y;
  size_t base = (size_t)b * per;
  float norm = sqrtf(nAcc[b]);
  float scale = fminf(1.0f, CLIP_C / (norm + EPS_C));
  float lrs = LR_C * scale;
  size_t stride = (size_t)gridDim.x * blockDim.x;
  for (size_t i = blockIdx.x * (size_t)blockDim.x + threadIdx.x; i < per; i += stride) {
    float w = Wf[base + i] + lrs * DELTA[base + i];
    Wf[base + i] = w;
    Wb[base + i] = f2bf(w);
  }
}

// ------------------------------- WMMA GEMMs --------------------------------
// 256 threads = 8 waves. Double-buffered LDS + async-to-LDS staging:
//   wait_asynccnt(0); barrier; issue asyncs into buf p^1; compute from buf p.

// Z = silu(x @ Wg^T) * (x @ Wu^T)   [16384 x 2688] bf16
// Block tile 128(M) x 64(N); waves 4x2; wave tile 32x32; dual accumulators.
__global__ __launch_bounds__(256) void k_gate(const bf16* __restrict__ X,
                                              const bf16* __restrict__ Wg,
                                              const bf16* __restrict__ Wu,
                                              bf16* __restrict__ Z) {
  __shared__ bf16 As[2][128 * LP];
  __shared__ bf16 Bgs[2][64 * LP];
  __shared__ bf16 Bus[2][64 * LP];
  int tid = threadIdx.x, lane = tid & 31, wid = tid >> 5;
  int wm = (wid >> 1) * 32, wn = (wid & 1) * 32;
  int rowBase = blockIdx.x * 128, colBase = blockIdx.y * 64;
  int r0 = tid >> 2, k8 = (tid & 3) << 3;

  const bf16* pA0 = X + (size_t)(rowBase + r0) * DIMK + k8;
  const bf16* pA1 = X + (size_t)(rowBase + r0 + 64) * DIMK + k8;
  const bf16* pBg = Wg + (size_t)(colBase + r0) * DIMK + k8;
  const bf16* pBu = Wu + (size_t)(colBase + r0) * DIMK + k8;
  unsigned dA0[2], dA1[2], dBg[2], dBu[2];
#pragma unroll
  for (int p = 0; p < 2; ++p) {
    dA0[p] = lds_off(&As[p][r0 * LP + k8]);
    dA1[p] = lds_off(&As[p][(r0 + 64) * LP + k8]);
    dBg[p] = lds_off(&Bgs[p][r0 * LP + k8]);
    dBu[p] = lds_off(&Bus[p][r0 * LP + k8]);
  }
  async_ld_b128(dA0[0], pA0);
  async_ld_b128(dA1[0], pA1);
  async_ld_b128(dBg[0], pBg);
  async_ld_b128(dBu[0], pBu);

  v8f accG[2][2] = {}, accU[2][2] = {};
  for (int k0 = 0; k0 < DIMK; k0 += BKT) {
    int p = (k0 >> 5) & 1;
    wait_asynccnt0();
    __syncthreads();
    if (k0 + BKT < DIMK) {
      async_ld_b128(dA0[p ^ 1], pA0 + k0 + BKT);
      async_ld_b128(dA1[p ^ 1], pA1 + k0 + BKT);
      async_ld_b128(dBg[p ^ 1], pBg + k0 + BKT);
      async_ld_b128(dBu[p ^ 1], pBu + k0 + BKT);
    }
    v16bf af[2], bgf[2], buf[2];
#pragma unroll
    for (int mi = 0; mi < 2; ++mi) af[mi] = lds_frag(As[p], wm + mi * 16, lane);
#pragma unroll
    for (int ni = 0; ni < 2; ++ni) {
      bgf[ni] = lds_frag(Bgs[p], wn + ni * 16, lane);
      buf[ni] = lds_frag(Bus[p], wn + ni * 16, lane);
    }
#pragma unroll
    for (int mi = 0; mi < 2; ++mi)
#pragma unroll
      for (int ni = 0; ni < 2; ++ni) {
        accG[mi][ni] = wmma_bf16(af[mi], bgf[ni], accG[mi][ni]);
        accU[mi][ni] = wmma_bf16(af[mi], buf[ni], accU[mi][ni]);
      }
  }
#pragma unroll
  for (int mi = 0; mi < 2; ++mi)
#pragma unroll
    for (int ni = 0; ni < 2; ++ni)
#pragma unroll
      for (int r = 0; r < 8; ++r) {
        int m = rowBase + wm + mi * 16 + ((lane >> 4) << 3) + r;
        int n = colBase + wn + ni * 16 + (lane & 15);
        float g = accG[mi][ni][r], u = accU[mi][ni][r];
        float z = g * __builtin_amdgcn_rcpf(1.0f + __expf(-g)) * u;  // silu(g)*u
        Z[(size_t)m * HIDDENK + n] = f2bf(z);
      }
}

// V[t] = M0 @ tok[t] + M1 @ tok[t+1 within chunk]   [16384 x 1024] f32
// Block tile 128 x 64; waves 4x2; wave tile 32x32; dual A operands.
// Zero rows (last token of each chunk) pre-written to both buffers once.
__global__ __launch_bounds__(256) void k_vgemm(const bf16* __restrict__ T,
                                               const bf16* __restrict__ M0,
                                               const bf16* __restrict__ M1,
                                               float* __restrict__ V) {
  __shared__ bf16 A0s[2][128 * LP];
  __shared__ bf16 A1s[2][128 * LP];
  __shared__ bf16 B0s[2][64 * LP];
  __shared__ bf16 B1s[2][64 * LP];
  int tid = threadIdx.x, lane = tid & 31, wid = tid >> 5;
  int wm = (wid >> 1) * 32, wn = (wid & 1) * 32;
  int rowBase = blockIdx.x * 128, colBase = blockIdx.y * 64;
  int r0 = tid >> 2, k8 = (tid & 3) << 3;
  int g0 = rowBase + r0, g1 = rowBase + r0 + 64;
  bool l0 = (g0 & (CHUNKK - 1)) == (CHUNKK - 1);
  bool l1 = (g1 & (CHUNKK - 1)) == (CHUNKK - 1);

  const bf16* pA0 = T + (size_t)g0 * DIMK + k8;
  const bf16* pA1 = T + (size_t)g1 * DIMK + k8;
  const bf16* pS0 = T + (size_t)(g0 + 1) * DIMK + k8;   // shifted rows
  const bf16* pS1 = T + (size_t)(g1 + 1) * DIMK + k8;
  const bf16* pB0 = M0 + (size_t)(colBase + r0) * DIMK + k8;
  const bf16* pB1 = M1 + (size_t)(colBase + r0) * DIMK + k8;
  unsigned dA0[2], dA1[2], dS0[2], dS1[2], dB0[2], dB1[2];
#pragma unroll
  for (int p = 0; p < 2; ++p) {
    dA0[p] = lds_off(&A0s[p][r0 * LP + k8]);
    dA1[p] = lds_off(&A0s[p][(r0 + 64) * LP + k8]);
    dS0[p] = lds_off(&A1s[p][r0 * LP + k8]);
    dS1[p] = lds_off(&A1s[p][(r0 + 64) * LP + k8]);
    dB0[p] = lds_off(&B0s[p][r0 * LP + k8]);
    dB1[p] = lds_off(&B1s[p][r0 * LP + k8]);
  }
  // zero rows stay zero across all K-steps (asyncs for them are masked off)
  v8bf zz = bfzero8();
  if (l0) { *(v8bf*)&A1s[0][r0 * LP + k8] = zz; *(v8bf*)&A1s[1][r0 * LP + k8] = zz; }
  if (l1) { *(v8bf*)&A1s[0][(r0 + 64) * LP + k8] = zz; *(v8bf*)&A1s[1][(r0 + 64) * LP + k8] = zz; }

  async_ld_b128(dA0[0], pA0);
  async_ld_b128(dA1[0], pA1);
  if (!l0) async_ld_b128(dS0[0], pS0);
  if (!l1) async_ld_b128(dS1[0], pS1);
  async_ld_b128(dB0[0], pB0);
  async_ld_b128(dB1[0], pB1);

  v8f acc[2][2] = {};
  for (int k0 = 0; k0 < DIMK; k0 += BKT) {
    int p = (k0 >> 5) & 1;
    wait_asynccnt0();
    __syncthreads();
    if (k0 + BKT < DIMK) {
      async_ld_b128(dA0[p ^ 1], pA0 + k0 + BKT);
      async_ld_b128(dA1[p ^ 1], pA1 + k0 + BKT);
      if (!l0) async_ld_b128(dS0[p ^ 1], pS0 + k0 + BKT);
      if (!l1) async_ld_b128(dS1[p ^ 1], pS1 + k0 + BKT);
      async_ld_b128(dB0[p ^ 1], pB0 + k0 + BKT);
      async_ld_b128(dB1[p ^ 1], pB1 + k0 + BKT);
    }
    v16bf a0f[2], a1f[2], b0f[2], b1f[2];
#pragma unroll
    for (int mi = 0; mi < 2; ++mi) {
      a0f[mi] = lds_frag(A0s[p], wm + mi * 16, lane);
      a1f[mi] = lds_frag(A1s[p], wm + mi * 16, lane);
    }
#pragma unroll
    for (int ni = 0; ni < 2; ++ni) {
      b0f[ni] = lds_frag(B0s[p], wn + ni * 16, lane);
      b1f[ni] = lds_frag(B1s[p], wn + ni * 16, lane);
    }
#pragma unroll
    for (int mi = 0; mi < 2; ++mi)
#pragma unroll
      for (int ni = 0; ni < 2; ++ni) {
        acc[mi][ni] = wmma_bf16(a0f[mi], b0f[ni], acc[mi][ni]);
        acc[mi][ni] = wmma_bf16(a1f[mi], b1f[ni], acc[mi][ni]);
      }
  }
#pragma unroll
  for (int mi = 0; mi < 2; ++mi)
#pragma unroll
    for (int ni = 0; ni < 2; ++ni)
#pragma unroll
      for (int r = 0; r < 8; ++r) {
        int m = rowBase + wm + mi * 16 + ((lane >> 4) << 3) + r;
        int n = colBase + wn + ni * 16 + (lane & 15);
        V[(size_t)m * DIMK + n] = acc[mi][ni][r];
      }
}

// out = Z_c @ w^T ; err = V - out (stored transposed bf16). Also zeros nAcc.
// Block tile 128 x 128; waves 4x2; wave tile 32x64 (2x4 frags).
__global__ __launch_bounds__(256) void k_apply(const bf16* __restrict__ Z,
                                               const bf16* __restrict__ W,
                                               const float* __restrict__ V,
                                               float* __restrict__ OUT,
                                               bf16* __restrict__ ERRT,
                                               float* __restrict__ nAcc, int cidx) {
  __shared__ bf16 As[2][128 * LP];
  __shared__ bf16 Bs[2][128 * LP];
  int b = blockIdx.z;
  if (blockIdx.x == 0 && blockIdx.y == 0 && threadIdx.x == 0) nAcc[b] = 0.f;
  const bf16* A  = Z + ((size_t)b * SEQK + (size_t)cidx * CHUNKK) * HIDDENK;
  const bf16* Bt = W + (size_t)b * DIMK * HIDDENK;
  int tid = threadIdx.x, lane = tid & 31, wid = tid >> 5;
  int wm = (wid >> 1) * 32, wn = (wid & 1) * 64;
  int rowBase = blockIdx.x * 128, colBase = blockIdx.y * 128;
  int r0 = tid >> 2, k8 = (tid & 3) << 3;

  const bf16* pA0 = A + (size_t)(rowBase + r0) * HIDDENK + k8;
  const bf16* pA1 = A + (size_t)(rowBase + r0 + 64) * HIDDENK + k8;
  const bf16* pB0 = Bt + (size_t)(colBase + r0) * HIDDENK + k8;
  const bf16* pB1 = Bt + (size_t)(colBase + r0 + 64) * HIDDENK + k8;
  unsigned dA0[2], dA1[2], dB0[2], dB1[2];
#pragma unroll
  for (int p = 0; p < 2; ++p) {
    dA0[p] = lds_off(&As[p][r0 * LP + k8]);
    dA1[p] = lds_off(&As[p][(r0 + 64) * LP + k8]);
    dB0[p] = lds_off(&Bs[p][r0 * LP + k8]);
    dB1[p] = lds_off(&Bs[p][(r0 + 64) * LP + k8]);
  }
  async_ld_b128(dA0[0], pA0);
  async_ld_b128(dA1[0], pA1);
  async_ld_b128(dB0[0], pB0);
  async_ld_b128(dB1[0], pB1);

  v8f acc[2][4] = {};
  for (int k0 = 0; k0 < HIDDENK; k0 += BKT) {
    int p = (k0 >> 5) & 1;
    wait_asynccnt0();
    __syncthreads();
    if (k0 + BKT < HIDDENK) {
      async_ld_b128(dA0[p ^ 1], pA0 + k0 + BKT);
      async_ld_b128(dA1[p ^ 1], pA1 + k0 + BKT);
      async_ld_b128(dB0[p ^ 1], pB0 + k0 + BKT);
      async_ld_b128(dB1[p ^ 1], pB1 + k0 + BKT);
    }
    v16bf af[2], bf_[4];
#pragma unroll
    for (int mi = 0; mi < 2; ++mi) af[mi] = lds_frag(As[p], wm + mi * 16, lane);
#pragma unroll
    for (int ni = 0; ni < 4; ++ni) bf_[ni] = lds_frag(Bs[p], wn + ni * 16, lane);
#pragma unroll
    for (int mi = 0; mi < 2; ++mi)
#pragma unroll
      for (int ni = 0; ni < 4; ++ni)
        acc[mi][ni] = wmma_bf16(af[mi], bf_[ni], acc[mi][ni]);
  }
#pragma unroll
  for (int mi = 0; mi < 2; ++mi)
#pragma unroll
    for (int ni = 0; ni < 4; ++ni)
#pragma unroll
      for (int r = 0; r < 8; ++r) {
        int t = rowBase + wm + mi * 16 + ((lane >> 4) << 3) + r;   // 0..255
        int n = colBase + wn + ni * 16 + (lane & 15);              // d
        size_t gr = (size_t)b * SEQK + (size_t)cidx * CHUNKK + t;
        float o = acc[mi][ni][r];
        OUT[gr * DIMK + n] = o;
        float e = V[gr * DIMK + n] - o;
        ERRT[((size_t)b * DIMK + n) * CHUNKK + t] = f2bf(e);
      }
}

// delta = err^T @ Z_c / CHUNK ; accumulates per-batch sum(delta^2) into nAcc.
// Block tile 128 x 128; waves 4x2; wave tile 32x64. B needs an element
// transpose, so this kernel keeps register-prefetch + scatter staging.
__global__ __launch_bounds__(256) void k_delta(const bf16* __restrict__ ERRT,
                                               const bf16* __restrict__ Z,
                                               float* __restrict__ DELTA,
                                               float* __restrict__ nAcc, int cidx) {
  __shared__ bf16 As[128 * LP];
  __shared__ bf16 Bs[128 * LP];
  __shared__ float red[256];
  int b = blockIdx.z;
  const bf16* A  = ERRT + (size_t)b * DIMK * CHUNKK;                         // [D x CHUNK]
  const bf16* Bz = Z + ((size_t)b * SEQK + (size_t)cidx * CHUNKK) * HIDDENK; // [CHUNK x H]
  int tid = threadIdx.x, lane = tid & 31, wid = tid >> 5;
  int wm = (wid >> 1) * 32, wn = (wid & 1) * 64;
  int rowBase = blockIdx.x * 128, colBase = blockIdx.y * 128;
  int r0 = tid >> 2, k8 = (tid & 3) << 3;
  int kk = tid >> 4, n8 = (tid & 15) << 3;

  const bf16* pA0 = A + (size_t)(rowBase + r0) * CHUNKK + k8;
  const bf16* pA1 = A + (size_t)(rowBase + r0 + 64) * CHUNKK + k8;
  const bf16* pB0 = Bz + (size_t)kk * HIDDENK + colBase + n8;
  const bf16* pB1 = Bz + (size_t)(kk + 16) * HIDDENK + colBase + n8;
  v8bf aR0 = *(const v8bf*)(pA0);
  v8bf aR1 = *(const v8bf*)(pA1);
  v8bf bR0 = *(const v8bf*)(pB0);
  v8bf bR1 = *(const v8bf*)(pB1);

  v8f acc[2][4] = {};
  for (int k0 = 0; k0 < CHUNKK; k0 += BKT) {
    *(v8bf*)&As[r0 * LP + k8]        = aR0;
    *(v8bf*)&As[(r0 + 64) * LP + k8] = aR1;
#pragma unroll
    for (int j = 0; j < 8; ++j) {                    // transpose scatter
      Bs[(n8 + j) * LP + kk]      = bR0[j];
      Bs[(n8 + j) * LP + kk + 16] = bR1[j];
    }
    __syncthreads();
    if (k0 + BKT < CHUNKK) {
      aR0 = *(const v8bf*)(pA0 + k0 + BKT);
      aR1 = *(const v8bf*)(pA1 + k0 + BKT);
      bR0 = *(const v8bf*)(pB0 + (size_t)(k0 + BKT) * HIDDENK);
      bR1 = *(const v8bf*)(pB1 + (size_t)(k0 + BKT) * HIDDENK);
    }
    v16bf af[2], bf_[4];
#pragma unroll
    for (int mi = 0; mi < 2; ++mi) af[mi] = lds_frag(As, wm + mi * 16, lane);
#pragma unroll
    for (int ni = 0; ni < 4; ++ni) bf_[ni] = lds_frag(Bs, wn + ni * 16, lane);
#pragma unroll
    for (int mi = 0; mi < 2; ++mi)
#pragma unroll
      for (int ni = 0; ni < 4; ++ni)
        acc[mi][ni] = wmma_bf16(af[mi], bf_[ni], acc[mi][ni]);
    __syncthreads();
  }
  float local = 0.f;
#pragma unroll
  for (int mi = 0; mi < 2; ++mi)
#pragma unroll
    for (int ni = 0; ni < 4; ++ni)
#pragma unroll
      for (int r = 0; r < 8; ++r) {
        int m = rowBase + wm + mi * 16 + ((lane >> 4) << 3) + r;   // d
        int n = colBase + wn + ni * 16 + (lane & 15);              // h
        float dv = acc[mi][ni][r] * (1.0f / CHUNKK);
        DELTA[((size_t)b * DIMK + m) * HIDDENK + n] = dv;
        local += dv * dv;
      }
  red[tid] = local;
  __syncthreads();
  for (int s = 128; s > 0; s >>= 1) {
    if (tid < s) red[tid] += red[tid + s];
    __syncthreads();
  }
  if (tid == 0) atomicAdd(&nAcc[b], red[0]);
}

// ------------------------------- host driver -------------------------------

extern "C" void kernel_launch(void* const* d_in, const int* in_sizes, int n_in,
                              void* d_out, int out_size, void* d_ws, size_t ws_size,
                              hipStream_t stream) {
  (void)in_sizes; (void)n_in; (void)out_size; (void)ws_size;
  const float* x    = (const float*)d_in[0];
  const float* tok  = (const float*)d_in[1];
  const float* Win  = (const float*)d_in[2];
  const float* wdn  = (const float*)d_in[3];
  const float* conv = (const float*)d_in[4];
  const float* Wt   = (const float*)d_in[5];
  float* out = (float*)d_out;

  char* ws = (char*)d_ws;
  size_t off = 0;
  auto take = [&](size_t bytes) {
    char* p = ws + off;
    off += (bytes + 255) & ~(size_t)255;
    return p;
  };
  bf16*  Z    = (bf16*)take((size_t)ROWSK * HIDDENK * 2);          // 88.1 MB
  float* V    = (float*)take((size_t)ROWSK * DIMK * 4);            // 67.1 MB
  float* Wf   = (float*)take((size_t)BATCHK * DIMK * HIDDENK * 4); // 44.0 MB
  bf16*  Wb   = (bf16*)take((size_t)BATCHK * DIMK * HIDDENK * 2);  // 22.0 MB
  float* nAcc = (float*)take(256);
  bf16*  M01  = (bf16*)take((size_t)2 * DIMK * DIMK * 2);          // 4.2 MB
  char*  scratch = ws + off;                                       // reused region
  bf16*  Winb = (bf16*)take((size_t)HID2K * DIMK * 2);             // 11.0 MB
  bf16*  Xb   = (bf16*)take((size_t)ROWSK * DIMK * 2);             // 33.6 MB
  bf16*  Tb   = (bf16*)take((size_t)ROWSK * DIMK * 2);             // 33.6 MB
  // Aliases: these live only during the scan, after Winb/Xb/Tb are dead.
  float* DELTA = (float*)scratch;   // 44.0 MB over Winb+Xb (44.5 MB)
  bf16*  ERRT  = Tb;                //  2.1 MB over Tb

  // 1) fp32 -> bf16 copies of activations/weights
  k_cvt_bf16<<<2048, 256, 0, stream>>>(x,   Xb,   ROWSK * DIMK / 4);
  k_cvt_bf16<<<2048, 256, 0, stream>>>(tok, Tb,   ROWSK * DIMK / 4);
  k_cvt_bf16<<<2048, 256, 0, stream>>>(Win, Winb, HID2K * DIMK / 4);

  // 2) fold conv(K=2)+W_target into M0, M1
  k_fold<<<dim3(DIMK / 16, DIMK / 16, 2), dim3(16, 16), 0, stream>>>(Wt, conv, M01);

  // 3) SwiGLU gate GEMM -> Z   (tiles: 128x64)
  k_gate<<<dim3(ROWSK / 128, HIDDENK / 64), 256, 0, stream>>>(
      Xb, Winb, Winb + (size_t)HIDDENK * DIMK, Z);

  // 4) conv-as-GEMM -> V       (tiles: 128x64)
  k_vgemm<<<dim3(ROWSK / 128, DIMK / 64), 256, 0, stream>>>(
      Tb, M01, M01 + (size_t)DIMK * DIMK, V);

  // 5) fast weights init (per-batch copies of w_down, f32 + bf16)
  k_init_w<<<dim3(1024, BATCHK), 256, 0, stream>>>(wdn, Wf, Wb);

  // 6) sequential chunk scan: apply -> delta(+norm) -> clipped update
  for (int c = 0; c < NCHUNKK; ++c) {
    k_apply<<<dim3(CHUNKK / 128, DIMK / 128, BATCHK), 256, 0, stream>>>(
        Z, Wb, V, out, ERRT, nAcc, c);
    k_delta<<<dim3(DIMK / 128, HIDDENK / 128, BATCHK), 256, 0, stream>>>(
        ERRT, Z, DELTA, nAcc, c);
    k_update<<<dim3(1024, BATCHK), 256, 0, stream>>>(DELTA, nAcc, Wf, Wb);
  }
}
